// MGCNN_3255585210962
// MI455X (gfx1250) — matrix operand order repaired and checked by reading, not compiled, not run on hardware
//
#include <hip/hip_runtime.h>
#include <hip/hip_bf16.h>
#include <math.h>

typedef __bf16 bf16;
typedef __attribute__((ext_vector_type(16))) __bf16 v16bf;
typedef __attribute__((ext_vector_type(8)))  float  v8f;
typedef unsigned int u32;
typedef __attribute__((ext_vector_type(4)))  u32    u32x4;
typedef __attribute__((ext_vector_type(4)))  unsigned int ui32x4;
typedef __attribute__((ext_vector_type(8)))  int    i32x8;
typedef __attribute__((ext_vector_type(4)))  int    i32x4;

#define LDIM  1024
#define PLANE (1024u*1024u)

#if __has_builtin(__builtin_amdgcn_tensor_load_to_lds)
#define USE_TDM 1
#else
#define USE_TDM 0
#endif

__device__ __forceinline__ float sigmoidf_(float v) { return 1.0f / (1.0f + __expf(-v)); }

#if USE_TDM
// ---------------------------------------------------------------------------
// TDM: DMA a 128(row) x 32(K) bf16 tile of row-major A into LDS, with HW
// padding of 4 dwords after every 16 dwords (one 64B row) so the LDS row
// stride is 40 halves (80B) -> conflict-free b128 fragment reads.
// D# layout per CDNA5 ISA 8.3/8.4 (group0: count/lds/global/type,
// group1: data_size, pad, tensor dims, tile dims, strides).
// ---------------------------------------------------------------------------
__device__ __forceinline__ void tdm_load_tileA(const bf16* A, int rowbase, int kb,
                                               unsigned lds_addr)
{
    unsigned long long ga =
        (unsigned long long)(A + (size_t)rowbase * LDIM + (size_t)kb * 32);
    ui32x4 g0 = { 1u,                                   // count=1, user mode
                  lds_addr,                             // LDS byte address
                  (unsigned)ga,                         // global_addr[31:0]
                  (unsigned)(ga >> 32) | (2u << 30) };  // global_addr[56:32] | type=2
    i32x8 g1 = { (int)((1u << 16)        // data_size = 2 bytes
                     | (1u << 20)        // pad_enable
                     | (3u << 22)        // pad_interval: every 16 dwords (64B row)
                     | (3u << 25)),      // pad_amount: 4 dwords (16B)
                 (int)(1024u << 16),     // tensor_dim0[15:0] (=1024) in [63:48]
                 (int)(1024u << 16),     // tensor_dim0[31:16]=0 | tensor_dim1[15:0]=1024
                 (int)(32u << 16),       // tensor_dim1[31:16]=0 | tile_dim0=32
                 128,                    // tile_dim1=128 | tile_dim2=0
                 1024,                   // tensor_dim0_stride[31:0] = 1024 elements
                 0, 0 };                 // stride hi / dim1_stride = 0
    i32x4 g2 = { 0, 0, 0, 0 };
    i32x4 g3 = { 0, 0, 0, 0 };
#if defined(__clang_major__) && __clang_major__ >= 23
    i32x8 g4 = { 0, 0, 0, 0, 0, 0, 0, 0 };
    __builtin_amdgcn_tensor_load_to_lds(g0, g1, g2, g3, g4, 0);
#else
    __builtin_amdgcn_tensor_load_to_lds(g0, g1, g2, g3, 0);
#endif
}
#endif

// ---------------------------------------------------------------------------
// 1024^3 bf16 GEMM, fp32 accumulate, v_wmma_f32_16x16x32_bf16.
//   O = A @ B   where Bt holds B in N-major form: Bt[n*1024 + k] = B[k][n]
// (all B operands here are symmetric matrices or pre-transposed x, so Bt is
//  just the plane itself). Block tile 128x128, 8 waves, wave tile 32x64
// (2x4 accumulators). A staged in double-buffered LDS via TDM (async DMA,
// overlapped with WMMA on the other buffer); B fragments are two contiguous
// global_load_b128 per lane straight from L2.
// EPI==0: O=(bf16)acc    EPI==1: O=(bf16)(2*acc - P2)  (Chebyshev recurrence)
// ---------------------------------------------------------------------------
template<int EPI>
__global__ __launch_bounds__(256)
void gemm1k_bf16(const bf16* __restrict__ A, const bf16* __restrict__ Bt,
                 const bf16* __restrict__ P2, bf16* __restrict__ O)
{
    __shared__ __align__(16) bf16 sA[2][128 * 40];   // padded rows: 40 halves

    const int t    = threadIdx.x;
    const int lane = t & 31;
    const int wave = t >> 5;
    const int wm   = wave & 3;    // 4 row-groups of 32 rows
    const int wn   = wave >> 2;   // 2 col-groups of 64 cols
    const int rowbase = blockIdx.y * 128;
    const int colbase = blockIdx.x * 128;

    v8f acc[2][4] = {};

#if USE_TDM
    const unsigned ldsA0 = (unsigned)(unsigned long long)&sA[0][0];
    if (wave == 0) tdm_load_tileA(A, rowbase, 0, ldsA0);
#endif

    for (int kb = 0; kb < 32; ++kb) {
        const int buf = kb & 1;

#if USE_TDM
        if (wave == 0) __builtin_amdgcn_s_wait_tensorcnt(0);
        __syncthreads();                                    // publish sA[buf]
        if (wave == 0 && kb + 1 < 32)                       // overlap next DMA
            tdm_load_tileA(A, rowbase, kb + 1, ldsA0 + (unsigned)((buf ^ 1) * 128 * 40 * 2));
#else
        #pragma unroll
        for (int cc = 0; cc < 2; ++cc) {
            int chunk = t + cc * 256;
            int ar = chunk >> 2, a8 = (chunk & 3) * 8;
            u32x4 d = *(const u32x4*)(A + (size_t)(rowbase + ar) * LDIM + kb * 32 + a8);
            *(u32x4*)(&sA[buf][ar * 40 + a8]) = d;
        }
        __syncthreads();
#endif

        // A fragments from LDS: lane r16 = row, klo selects the ISA's K chunks
        union { v16bf v; u32x4 q[2]; } fa[2], fb[4];
        const int r16 = lane & 15;
        const int klo = (lane >> 4) * 8;
        #pragma unroll
        for (int mt = 0; mt < 2; ++mt) {
            const bf16* p = &sA[buf][(wm * 32 + mt * 16 + r16) * 40];
            fa[mt].q[0] = *(const u32x4*)(p + klo);
            fa[mt].q[1] = *(const u32x4*)(p + 16 + klo);
        }
        // B fragments direct from global (N-major): K chunks bo..bo+7 and 16+bo..16+bo+7 per lane
        const int bo = (lane >> 4) * 8;
        #pragma unroll
        for (int nt = 0; nt < 4; ++nt) {
            const bf16* p = Bt + (size_t)(colbase + wn * 64 + nt * 16 + r16) * LDIM
                               + kb * 32 + bo;
            fb[nt].q[0] = *(const u32x4*)(p);
            fb[nt].q[1] = *(const u32x4*)(p + 16);
        }
        #pragma unroll
        for (int mt = 0; mt < 2; ++mt)
            #pragma unroll
            for (int nt = 0; nt < 4; ++nt)
                acc[mt][nt] = __builtin_amdgcn_wmma_f32_16x16x32_bf16(
                    false, fa[mt].v, false, fb[nt].v, (short)0, acc[mt][nt], false, false);
        __syncthreads();                                    // done reading sA[buf]
    }

    // Epilogue. C layout: lanes 0-15 N=lane, M=r; lanes 16-31 N=lane-16, M=8+r.
    const int r16 = lane & 15;
    const int rhi = (lane >> 4) * 8;
    #pragma unroll
    for (int mt = 0; mt < 2; ++mt) {
        #pragma unroll
        for (int nt = 0; nt < 4; ++nt) {
            const int col = colbase + wn * 64 + nt * 16 + r16;
            #pragma unroll
            for (int r = 0; r < 8; ++r) {
                const int row = rowbase + wm * 32 + mt * 16 + rhi + r;
                const size_t idx = (size_t)row * LDIM + col;
                float a = acc[mt][nt][r];
                if (EPI == 1) a = 2.0f * a - (float)P2[idx];
                O[idx] = (bf16)a;
            }
        }
    }
}

// ---------------------------------------------------------------------------
// 1024x1024 bf16 transpose (for the non-symmetric B operand x)
// ---------------------------------------------------------------------------
__global__ __launch_bounds__(256)
void transpose_bf16(const bf16* __restrict__ in, bf16* __restrict__ out)
{
    __shared__ bf16 tile[32][33];
    const int bx = blockIdx.x * 32, by = blockIdx.y * 32;
    const int tx = threadIdx.x & 31, ty = threadIdx.x >> 5;   // 32 x 8
    #pragma unroll
    for (int r = 0; r < 4; ++r)
        tile[ty + r * 8][tx] = in[(size_t)(by + ty + r * 8) * LDIM + bx + tx];
    __syncthreads();
    #pragma unroll
    for (int r = 0; r < 4; ++r)
        out[(size_t)(bx + ty + r * 8) * LDIM + by + tx] = tile[tx][ty + r * 8];
}

// ---------------------------------------------------------------------------
// xc[p][o] = sum_t theta[t][o] * bf[t][p] + bias[o]   (t = i*5+j, 25 planes)
// ---------------------------------------------------------------------------
__global__ __launch_bounds__(256)
void conv_mix(const bf16* __restrict__ bfall, const float* __restrict__ theta,
              const float* __restrict__ bias, bf16* __restrict__ xc)
{
    __shared__ float sth[25 * 32];
    __shared__ float sb[32];
    for (int i = threadIdx.x; i < 25 * 32; i += 256) sth[i] = theta[i];
    if (threadIdx.x < 32) sb[threadIdx.x] = bias[threadIdx.x];
    __syncthreads();

    const unsigned p = blockIdx.x * 256 + threadIdx.x;
    float accv[32];
    #pragma unroll
    for (int o = 0; o < 32; ++o) accv[o] = sb[o];
    for (int tt = 0; tt < 25; ++tt) {
        const float v = (float)bfall[(size_t)tt * PLANE + p];
        #pragma unroll
        for (int o = 0; o < 32; ++o) accv[o] += v * sth[tt * 32 + o];
    }
    union { u32x4 q[4]; bf16 h[32]; } outp;
    #pragma unroll
    for (int o = 0; o < 32; ++o) outp.h[o] = (bf16)accv[o];
    u32x4* dst = (u32x4*)(xc + (size_t)p * 32);
    #pragma unroll
    for (int qq = 0; qq < 4; ++qq) dst[qq] = outp.q[qq];
}

// ---------------------------------------------------------------------------
// Fused LSTM step: 16 bf16 WMMAs per wave compute all four gates for 16 pixels
// (K=32, N=32 as two 16-wide tiles, fp32 accumulate), then cell/hidden update
// and x += tanh(c @ W_out + b_out) via wave32 shfl reduction.
// ---------------------------------------------------------------------------
__global__ __launch_bounds__(256)
void lstm_step(const bf16* __restrict__ xc, bf16* __restrict__ h, float* __restrict__ c,
               const bf16* __restrict__ Wpk,
               const float* __restrict__ b_f, const float* __restrict__ b_i,
               const float* __restrict__ b_o, const float* __restrict__ b_c,
               const float* __restrict__ Wout, const float* __restrict__ bout,
               float* __restrict__ x)
{
    const int lane = threadIdx.x & 31;
    const int wave = threadIdx.x >> 5;
    const int pix0 = blockIdx.x * 128 + wave * 16;
    const int r16  = lane & 15;
    const int hi   = lane >> 4;

    union { v16bf v; u32x4 q[2]; } axc, ah;
    {
        const int klo = hi * 8;
        const bf16* p  = xc + (size_t)(pix0 + r16) * 32;
        axc.q[0] = *(const u32x4*)(p + klo);
        axc.q[1] = *(const u32x4*)(p + 16 + klo);
        const bf16* ph = h + (size_t)(pix0 + r16) * 32;
        ah.q[0] = *(const u32x4*)(ph + klo);
        ah.q[1] = *(const u32x4*)(ph + 16 + klo);
    }

    const float* gb[4] = { b_f, b_i, b_o, b_c };
    v8f gates[4][2];
    const int bk0 = hi * 8;
    #pragma unroll
    for (int g = 0; g < 4; ++g) {
        #pragma unroll
        for (int nt = 0; nt < 2; ++nt) {
            union { v16bf v; u32x4 q[2]; } bw, bu;
            const bf16* pw = Wpk + (size_t)(2 * g) * 1024 + (nt * 16 + r16) * 32 + bk0;
            bw.q[0] = *(const u32x4*)pw;       bw.q[1] = *(const u32x4*)(pw + 16);
            const bf16* pu = Wpk + (size_t)(2 * g + 1) * 1024 + (nt * 16 + r16) * 32 + bk0;
            bu.q[0] = *(const u32x4*)pu;       bu.q[1] = *(const u32x4*)(pu + 16);
            v8f a = {};
            a = __builtin_amdgcn_wmma_f32_16x16x32_bf16(false, axc.v, false, bw.v, (short)0, a, false, false);
            a = __builtin_amdgcn_wmma_f32_16x16x32_bf16(false, ah.v,  false, bu.v, (short)0, a, false, false);
            gates[g][nt] = a;
        }
    }

    float pdx[8];
    #pragma unroll
    for (int r = 0; r < 8; ++r) pdx[r] = 0.0f;

    #pragma unroll
    for (int nt = 0; nt < 2; ++nt) {
        const int ch = nt * 16 + r16;
        const float bfv = gb[0][ch], biv = gb[1][ch], bov = gb[2][ch], bcv = gb[3][ch];
        const float wo  = Wout[ch];
        #pragma unroll
        for (int r = 0; r < 8; ++r) {
            const int pix = pix0 + hi * 8 + r;
            const float fv = sigmoidf_(gates[0][nt][r] + bfv);
            const float iv = sigmoidf_(gates[1][nt][r] + biv);
            const float ov = sigmoidf_(gates[2][nt][r] + bov);
            const float gv = sigmoidf_(gates[3][nt][r] + bcv);
            const size_t idx = (size_t)pix * 32 + ch;
            const float cn = fv * c[idx] + iv * gv;
            c[idx] = cn;
            h[idx] = (bf16)(ov * sigmoidf_(cn));   // reference: h = o * sigmoid(c)
            pdx[r] += cn * wo;
        }
    }
    #pragma unroll
    for (int r = 0; r < 8; ++r) {
        float s = pdx[r];
        s += __shfl_xor(s, 1, 32);
        s += __shfl_xor(s, 2, 32);
        s += __shfl_xor(s, 4, 32);
        s += __shfl_xor(s, 8, 32);
        if (r16 == 0) {
            const int pix = pix0 + hi * 8 + r;
            x[pix] = x[pix] + tanhf(s + bout[0]);
        }
    }
}

// ---------------------------------------------------------------------------
__global__ void pack_weights(const float* W_f, const float* U_f, const float* W_i, const float* U_i,
                             const float* W_o, const float* U_o, const float* W_c, const float* U_c,
                             bf16* Wpk)
{
    const int t = blockIdx.x * 256 + threadIdx.x;
    if (t >= 8 * 1024) return;
    const int m = t >> 10, rem = t & 1023, n = rem >> 5, k = rem & 31;
    const float* src;
    switch (m) {
        case 0: src = W_f; break; case 1: src = U_f; break;
        case 2: src = W_i; break; case 3: src = U_i; break;
        case 4: src = W_o; break; case 5: src = U_o; break;
        case 6: src = W_c; break; default: src = U_c; break;
    }
    Wpk[t] = (bf16)src[k * 32 + n];
}

__global__ void cvt_f32_bf16(const float* __restrict__ in, bf16* __restrict__ out, int n)
{
    const int i = blockIdx.x * 256 + threadIdx.x;
    if (i < n) out[i] = (bf16)in[i];
}

__global__ void set_identity(bf16* __restrict__ Tr0, bf16* __restrict__ Tc0)
{
    const int i = blockIdx.x * 256 + threadIdx.x;
    const bf16 v = (bf16)(((i >> 10) == (i & 1023)) ? 1.0f : 0.0f);
    Tr0[i] = v; Tc0[i] = v;
}

// ---------------------------------------------------------------------------
extern "C" void kernel_launch(void* const* d_in, const int* in_sizes, int n_in,
                              void* d_out, int out_size, void* d_ws, size_t ws_size,
                              hipStream_t stream)
{
    const float* x_in  = (const float*)d_in[0];
    const float* Lr    = (const float*)d_in[1];
    const float* Lc    = (const float*)d_in[2];
    const float* theta = (const float*)d_in[3];
    const float* bias  = (const float*)d_in[4];
    const float* W_f = (const float*)d_in[5],  *U_f = (const float*)d_in[6],  *b_f = (const float*)d_in[7];
    const float* W_i = (const float*)d_in[8],  *U_i = (const float*)d_in[9],  *b_i = (const float*)d_in[10];
    const float* W_o = (const float*)d_in[11], *U_o = (const float*)d_in[12], *b_o = (const float*)d_in[13];
    const float* W_c = (const float*)d_in[14], *U_c = (const float*)d_in[15], *b_c = (const float*)d_in[16];
    const float* Wout = (const float*)d_in[17], *bout = (const float*)d_in[18];

    char* ws = (char*)d_ws;
    size_t off = 0;
    auto alloc = [&](size_t bytes) -> char* {
        char* p = ws + off;
        off += (bytes + 255) & ~(size_t)255;
        return p;
    };
    bf16*  Trbf  = (bf16*)alloc((size_t)5 * PLANE * 2);    // Chebyshev basis rows (symmetric)
    bf16*  Tcbf  = (bf16*)alloc((size_t)5 * PLANE * 2);    // Chebyshev basis cols (symmetric)
    bf16*  xbf   = (bf16*)alloc((size_t)PLANE * 2);        // x bf16
    bf16*  xbfT  = (bf16*)alloc((size_t)PLANE * 2);        // x^T bf16 (B operand form)
    bf16*  xrbf  = (bf16*)alloc((size_t)5 * PLANE * 2);    // Tr[i] @ x
    bf16*  bfall = (bf16*)alloc((size_t)25 * PLANE * 2);   // xr[i] @ Tc[j]
    bf16*  xcbf  = (bf16*)alloc((size_t)PLANE * 32 * 2);   // conv output per pixel
    bf16*  hbf   = (bf16*)alloc((size_t)PLANE * 32 * 2);   // hidden state
    float* cst   = (float*)alloc((size_t)PLANE * 32 * 4);  // cell state (fp32)
    float* xf    = (float*)alloc((size_t)PLANE * 4);       // fp32 master x
    bf16*  Wpk   = (bf16*)alloc((size_t)8 * 1024 * 2);     // packed gate weights

    hipMemcpyAsync(xf, x_in, (size_t)PLANE * 4, hipMemcpyDeviceToDevice, stream);
    hipMemsetAsync(hbf, 0, (size_t)PLANE * 32 * 2, stream);
    hipMemsetAsync(cst, 0, (size_t)PLANE * 32 * 4, stream);

    const dim3 blk(256);
    const dim3 ggrid(8, 8);
    const dim3 tgrid(32, 32);

    cvt_f32_bf16<<<PLANE / 256, blk, 0, stream>>>(Lr, Trbf + PLANE, PLANE);
    cvt_f32_bf16<<<PLANE / 256, blk, 0, stream>>>(Lc, Tcbf + PLANE, PLANE);
    set_identity<<<PLANE / 256, blk, 0, stream>>>(Trbf, Tcbf);
    pack_weights<<<32, blk, 0, stream>>>(W_f, U_f, W_i, U_i, W_o, U_o, W_c, U_c, Wpk);

    // T_k = 2 * L @ T_{k-1} - T_{k-2}   (all operands symmetric -> B as-is)
    for (int k = 2; k <= 4; ++k) {
        gemm1k_bf16<1><<<ggrid, blk, 0, stream>>>(Trbf + PLANE, Trbf + (size_t)(k - 1) * PLANE,
                                                  Trbf + (size_t)(k - 2) * PLANE, Trbf + (size_t)k * PLANE);
        gemm1k_bf16<1><<<ggrid, blk, 0, stream>>>(Tcbf + PLANE, Tcbf + (size_t)(k - 1) * PLANE,
                                                  Tcbf + (size_t)(k - 2) * PLANE, Tcbf + (size_t)k * PLANE);
    }

    for (int it = 0; it < 3; ++it) {
        cvt_f32_bf16<<<PLANE / 256, blk, 0, stream>>>(xf, xbf, PLANE);
        transpose_bf16<<<tgrid, blk, 0, stream>>>(xbf, xbfT);
        // xr[i] = Tr[i] @ x      (B = x, not symmetric -> use x^T)
        for (int i = 0; i < 5; ++i)
            gemm1k_bf16<0><<<ggrid, blk, 0, stream>>>(Trbf + (size_t)i * PLANE, xbfT,
                                                      nullptr, xrbf + (size_t)i * PLANE);
        // bf[i,j] = xr[i] @ Tc[j] (Tc symmetric -> B as-is)
        for (int i = 0; i < 5; ++i)
            for (int j = 0; j < 5; ++j)
                gemm1k_bf16<0><<<ggrid, blk, 0, stream>>>(xrbf + (size_t)i * PLANE,
                                                          Tcbf + (size_t)j * PLANE,
                                                          nullptr,
                                                          bfall + (size_t)(i * 5 + j) * PLANE);
        conv_mix<<<PLANE / 256, blk, 0, stream>>>(bfall, theta, bias, xcbf);
        lstm_step<<<PLANE / 128, blk, 0, stream>>>(xcbf, hbf, cst, Wpk,
                                                   b_f, b_i, b_o, b_c, Wout, bout, xf);
    }

    hipMemcpyAsync(d_out, xf, (size_t)PLANE * 4, hipMemcpyDeviceToDevice, stream);
}
